// ResidualVectorQuantizer_76544907149317
// MI455X (gfx1250) — compile-verified, matrix operand order checked by reading, not run on hardware
//
#include <hip/hip_runtime.h>
#include <math.h>

// Problem constants (from reference)
constexpr int kB = 8, kS = 4096, kD = 512;
constexpr int kQ = 8, kK = 1024;
constexpr int kN = kB * kS;               // 32768 rows

typedef __attribute__((ext_vector_type(16))) __bf16        v16bf;
typedef __attribute__((ext_vector_type(8)))  float         v8f;
typedef __attribute__((ext_vector_type(8)))  unsigned int  v8u;

// ---- bf16 helpers (round-to-nearest-even) ----
__device__ __forceinline__ unsigned int bfbits(float f) {
    unsigned int u = __float_as_uint(f);
    return (u + 0x7fffu + ((u >> 16) & 1u)) >> 16;
}
__device__ __forceinline__ unsigned int pk2(float lo, float hi) {
    return (bfbits(hi) << 16) | bfbits(lo);
}
// Convert 8 consecutive f32 -> 8 packed bf16 (one 16B uint4)
__device__ __forceinline__ uint4 cvt8(const float* p) {
    float4 a = *(const float4*)p;
    float4 b = *(const float4*)(p + 4);
    uint4 r;
    r.x = pk2(a.x, a.y); r.y = pk2(a.z, a.w);
    r.z = pk2(b.x, b.y); r.w = pk2(b.z, b.w);
    return r;
}
// Assemble a 16x32 bf16 WMMA fragment from two 16B runs
__device__ __forceinline__ v16bf mkfrag(uint4 lo, uint4 hi) {
    v8u t;
    t[0] = lo.x; t[1] = lo.y; t[2] = lo.z; t[3] = lo.w;
    t[4] = hi.x; t[5] = hi.y; t[6] = hi.z; t[7] = hi.w;
    return __builtin_bit_cast(v16bf, t);
}

// ---- prep: convert codebooks to bf16 ----
__global__ void prep_cb_kernel(const float* __restrict__ cb,
                               unsigned short* __restrict__ cbh, int n) {
    for (int i = blockIdx.x * blockDim.x + threadIdx.x; i < n;
         i += gridDim.x * blockDim.x)
        cbh[i] = (unsigned short)bfbits(cb[i]);
}

// ---- prep: ||c||^2 per codebook entry ----
__global__ void prep_norm_kernel(const float* __restrict__ cb,
                                 float* __restrict__ cnorm) {
    int i = blockIdx.x * blockDim.x + threadIdx.x;   // 0..Q*K
    if (i >= kQ * kK) return;
    const float4* p = (const float4*)(cb + (size_t)i * kD);
    float s = 0.f;
    for (int j = 0; j < kD / 4; ++j) {
        float4 v = p[j];
        s += v.x * v.x + v.y * v.y + v.z * v.z + v.w * v.w;
    }
    cnorm[i] = s;
}

// ---- main per-step kernel: distances (WMMA) + argmin + residual update ----
// Each WG: 128 rows; each of 8 waves owns a 16-row stripe.
// A fragments (row stripe of residual, bf16) stay resident in registers and
// are reused across all 64 codebook column chunks. The argmin is tracked
// per-lane inside the chunk loop (cheap: 8 cmp-selects per chunk) and the
// 16-lane butterfly reduction runs exactly once at the end, so the hot loop
// issue stream is dominated by WMMA + B-streaming loads.
__global__ __launch_bounds__(256, 1)
void rvq_step_kernel(int q,
                     const float* asrc,                 // residual source (x when q==0); may alias res
                     float*       res,                  // residual destination
                     const float* __restrict__ cbf,     // f32 codebooks [Q,K,D]
                     const unsigned short* __restrict__ cbh,   // bf16 codebooks [Q,K,D]
                     const float* __restrict__ cnorm,   // [Q,K]
                     int* __restrict__ outIdx)          // [N,Q]
{
    const int lane  = threadIdx.x & 31;
    const int wave  = threadIdx.x >> 5;
    const int half  = lane >> 4;          // 0: rows v, 1: rows v+8 (C layout)
    const int ln    = lane & 15;
    const int koff  = half * 8;           // A/B fragment K offset per lane half
    const int rowBase = blockIdx.x * 128 + wave * 16;
    const int myRow   = rowBase + ln;     // row this lane feeds into A fragments

    // --- Load + convert A fragments: 16 x (16x32 bf16), resident in VGPRs ---
    v16bf a[16];
    const float* ap = asrc + (size_t)myRow * kD;
    #pragma unroll
    for (int f = 0; f < 16; ++f) {
        int d0 = 32 * f + koff;
        a[f] = mkfrag(cvt8(ap + d0), cvt8(ap + d0 + 16));
    }

    const unsigned short* bq  = cbh + (size_t)q * kK * kD;
    const float*          cnq = cnorm + q * kK;

    // Per-lane running best over the 64 columns this lane observes per row.
    float bestV[8];
    int   bestI[8];
    #pragma unroll
    for (int v = 0; v < 8; ++v) { bestV[v] = 3.4e38f; bestI[v] = 0x7fffffff; }

    // --- Sweep 64 chunks of 16 codebook columns ---
    for (int c = 0; c < kK / 16; ++c) {
        const int colBase = c * 16;
        const unsigned short* bp = bq + (size_t)(colBase + ln) * kD;

        v8f acc = {};
        #pragma unroll
        for (int f = 0; f < 16; ++f) {
            int d0 = 32 * f + koff;
            uint4 lo = *(const uint4*)(bp + d0);
            uint4 hi = *(const uint4*)(bp + d0 + 16);
            v16bf b  = mkfrag(lo, hi);
            acc = __builtin_amdgcn_wmma_f32_16x16x32_bf16(
                      false, a[f], false, b, (short)0, acc, false, false);
        }

        const float cn  = cnq[colBase + ln];
        const int   idx = colBase + ln;
        #pragma unroll
        for (int v = 0; v < 8; ++v) {
            // dist (minus the row-constant ||f||^2) for (row v+8*half, col)
            float val = __builtin_fmaf(-2.0f, acc[v], cn);
            // strict < keeps the earliest (lowest-index) minimum per lane
            if (val < bestV[v]) { bestV[v] = val; bestI[v] = idx; }
        }
    }

    // --- One-time min+argmin across the 16 lanes of each half ---
    #pragma unroll
    for (int v = 0; v < 8; ++v) {
        float val = bestV[v];
        int   idx = bestI[v];
        #pragma unroll
        for (int m = 8; m >= 1; m >>= 1) {
            float ov = __shfl_xor(val, m, 32);
            int   oi = __shfl_xor(idx, m, 32);
            if (ov < val || (ov == val && oi < idx)) { val = ov; idx = oi; }
        }
        bestV[v] = val;
        bestI[v] = idx;
    }

    // --- Write indices: lane ln==v writes row rowBase + v + 8*half ---
    #pragma unroll
    for (int v = 0; v < 8; ++v) {
        if (ln == v) {
            int row = rowBase + v + 8 * half;
            outIdx[(size_t)row * kQ + q] = bestI[v];
        }
    }

    // --- Residual update: res[row] = asrc[row] - cb[q][best[row]] ---
    const float* cq = cbf + (size_t)q * kK * kD;
    #pragma unroll
    for (int lr = 0; lr < 16; ++lr) {
        const int v = lr & 7;
        const int srcLane = (lr < 8) ? v : (16 + v);
        const int ci  = __shfl(bestI[v], srcLane, 32);
        const int row = rowBase + lr;
        const float4* xs = (const float4*)(asrc + (size_t)row * kD) + lane * 4;
        const float4* cs = (const float4*)(cq   + (size_t)ci  * kD) + lane * 4;
        float4*       rs = (float4*)      (res  + (size_t)row * kD) + lane * 4;
        #pragma unroll
        for (int j = 0; j < 4; ++j) {
            float4 xv = xs[j], cv = cs[j];
            float4 o;
            o.x = xv.x - cv.x; o.y = xv.y - cv.y;
            o.z = xv.z - cv.z; o.w = xv.w - cv.w;
            rs[j] = o;
        }
    }
}

// ---- finalize: quantized = x - residual (in place: residual lives in outQ) ----
__global__ void finalize_kernel(const float* __restrict__ x, float* outq, int n) {
    for (int i = blockIdx.x * blockDim.x + threadIdx.x; i < n;
         i += gridDim.x * blockDim.x)
        outq[i] = x[i] - outq[i];
}

// ---- diversity loss: single wave ----
__global__ void losses_kernel(const float* __restrict__ usage, float* out) {
    const int lane = threadIdx.x;
    float esum = 0.f;
    for (int q = 0; q < kQ; ++q) {
        const float* u = usage + q * kK;
        float s = 0.f;
        for (int k = lane; k < kK; k += 32) s += u[k];
        #pragma unroll
        for (int m = 16; m >= 1; m >>= 1) s += __shfl_xor(s, m, 32);
        const float inv = 1.f / (s + 1e-5f);
        float e = 0.f;
        for (int k = lane; k < kK; k += 32) {
            float p = u[k] * inv;
            e -= p * logf(p + 1e-10f);
        }
        #pragma unroll
        for (int m = 16; m >= 1; m >>= 1) e += __shfl_xor(e, m, 32);
        esum += e;
    }
    if (lane == 0) {
        float div = 1.f - (esum / (float)kQ) / logf((float)kK);
        out[0] = 0.f;          // commitment loss (eval mode)
        out[1] = div;          // diversity loss
        out[2] = 0.1f * div;   // total vq loss
    }
}

extern "C" void kernel_launch(void* const* d_in, const int* in_sizes, int n_in,
                              void* d_out, int out_size, void* d_ws, size_t ws_size,
                              hipStream_t stream) {
    const float* x     = (const float*)d_in[0];   // [B,S,D]
    const float* cb    = (const float*)d_in[1];   // [Q,K,D]
    const float* usage = (const float*)d_in[2];   // [Q,K]

    float* outQ    = (float*)d_out;                              // quantized [N,D]
    int*   outIdx  = (int*)  (outQ + (size_t)kN * kD);           // indices [N,Q]
    float* outLoss = outQ + (size_t)kN * kD + (size_t)kN * kQ;   // 3 scalars

    unsigned short* cbh   = (unsigned short*)d_ws;                       // bf16 codebooks (8MB)
    float*          cnorm = (float*)((char*)d_ws + (size_t)kQ * kK * kD * 2);

    // residual buffer aliases the quantized output region
    float* res = outQ;

    prep_cb_kernel  <<<2048, 256, 0, stream>>>(cb, cbh, kQ * kK * kD);
    prep_norm_kernel<<<(kQ * kK + 255) / 256, 256, 0, stream>>>(cb, cnorm);

    for (int q = 0; q < kQ; ++q) {
        rvq_step_kernel<<<kN / 128, 256, 0, stream>>>(
            q, (q == 0 ? x : (const float*)res), res, cb, cbh, cnorm, outIdx);
    }

    finalize_kernel<<<2048, 256, 0, stream>>>(x, outQ, kN * kD);
    losses_kernel  <<<1, 32, 0, stream>>>(usage, outLoss);
}
